// AttnBlock_25469156065652
// MI455X (gfx1250) — matrix operand order repaired and checked by reading, not compiled, not run on hardware
//
#include <hip/hip_runtime.h>

typedef __attribute__((ext_vector_type(16))) __bf16   v16bf;
typedef __attribute__((ext_vector_type(8)))  float    v8f;
typedef __attribute__((ext_vector_type(4)))  unsigned u32x4;

#define CCH 512
#define NSP 1024
#define NB  16

union FragU { v16bf v; u32x4 q[2]; };

// ---------------------------------------------------------------------------
// CDNA5 async global->LDS copy (16B per lane), tracked by ASYNCcnt.
// ---------------------------------------------------------------------------
__device__ __forceinline__ void async_b128(unsigned lds_off, const void* g) {
  asm volatile("global_load_async_to_lds_b128 %0, %1, off"
               :: "v"(lds_off), "v"((unsigned long long)(uintptr_t)g)
               : "memory");
}
__device__ __forceinline__ void wait_async0() {
  asm volatile("s_wait_asynccnt 0" ::: "memory");
}
__device__ __forceinline__ unsigned lds_off32(const void* p) {
  return (unsigned)(uintptr_t)p;  // LDS generic addr: low 32 bits = LDS offset
}

// ---------------------------------------------------------------------------
// GroupNorm: one block per (batch, group). Writes rT[b][n][c] (K-major for
// the QKV GEMM B-operand): 16 contiguous bf16 channels per pixel.
// ---------------------------------------------------------------------------
__global__ void gn_kernel(const float* __restrict__ x,
                          const float* __restrict__ gw,
                          const float* __restrict__ gb,
                          __bf16* __restrict__ rT) {
  const int t = threadIdx.x;
  const int b = blockIdx.x >> 5;
  const int g = blockIdx.x & 31;
  const size_t base = ((size_t)b * CCH + (size_t)g * 16) * NSP;
  const float* xb = x + base;
  float s = 0.f, ss = 0.f;
  for (int i = t; i < 16 * NSP; i += 256) {
    const float v = xb[i];
    s += v; ss += v * v;
  }
  __shared__ float rs[256], rq[256];
  rs[t] = s; rq[t] = ss;
  __syncthreads();
  for (int off = 128; off > 0; off >>= 1) {
    if (t < off) { rs[t] += rs[t + off]; rq[t] += rq[t + off]; }
    __syncthreads();
  }
  const float inv  = 1.f / 16384.f;
  const float mean = rs[0] * inv;
  const float var  = rq[0] * inv - mean * mean;
  const float rstd = rsqrtf(var + 1e-6f);
  const int cbase = g << 4;
  float scl[16], sft[16];
#pragma unroll
  for (int cc = 0; cc < 16; cc++) {
    scl[cc] = gw[cbase + cc] * rstd;
    sft[cc] = gb[cbase + cc] - mean * scl[cc];
  }
#pragma unroll
  for (int rep = 0; rep < 4; rep++) {
    const int n = rep * 256 + t;
    union { __bf16 h[16]; u32x4 q[2]; } u;
#pragma unroll
    for (int cc = 0; cc < 16; cc++)
      u.h[cc] = (__bf16)(xb[cc * NSP + n] * scl[cc] + sft[cc]);
    u32x4* dst = (u32x4*)(rT + ((size_t)b * NSP + n) * CCH + cbase);
    dst[0] = u.q[0];
    dst[1] = u.q[1];
  }
}

// ---------------------------------------------------------------------------
// Convert weights to bf16 (no transposes needed with K-major B convention).
// ---------------------------------------------------------------------------
__global__ void wconv_kernel(const float* __restrict__ wq, const float* __restrict__ wk,
                             const float* __restrict__ wv, const float* __restrict__ wp,
                             __bf16* __restrict__ wqb, __bf16* __restrict__ wkb,
                             __bf16* __restrict__ wvb, __bf16* __restrict__ wpb) {
  const int i = blockIdx.x * 256 + threadIdx.x;
  if (i >= CCH * CCH) return;
  wqb[i] = (__bf16)wq[i];
  wkb[i] = (__bf16)wk[i];
  wvb[i] = (__bf16)wv[i];
  wpb[i] = (__bf16)wp[i];
}

// ---------------------------------------------------------------------------
// Row softmax over 1024 f32 -> bf16. One block (256 thr) per row.
// ---------------------------------------------------------------------------
__global__ void softmax_kernel(const float* __restrict__ S, __bf16* __restrict__ W) {
  const int t = threadIdx.x;
  const size_t base = (size_t)blockIdx.x * NSP;
  const float4 f = ((const float4*)(S + base))[t];
  float m = fmaxf(fmaxf(f.x, f.y), fmaxf(f.z, f.w));
  __shared__ float red[256];
  red[t] = m;
  __syncthreads();
  for (int off = 128; off > 0; off >>= 1) {
    if (t < off) red[t] = fmaxf(red[t], red[t + off]);
    __syncthreads();
  }
  m = red[0];
  __syncthreads();
  const float e0 = __expf(f.x - m), e1 = __expf(f.y - m);
  const float e2 = __expf(f.z - m), e3 = __expf(f.w - m);
  red[t] = e0 + e1 + e2 + e3;
  __syncthreads();
  for (int off = 128; off > 0; off >>= 1) {
    if (t < off) red[t] += red[t + off];
    __syncthreads();
  }
  const float rinv = 1.f / red[0];
  __bf16* wo = W + base + t * 4;
  wo[0] = (__bf16)(e0 * rinv);
  wo[1] = (__bf16)(e1 * rinv);
  wo[2] = (__bf16)(e2 * rinv);
  wo[3] = (__bf16)(e3 * rinv);
}

// ---------------------------------------------------------------------------
// bf16 WMMA GEMM:  D = A[MxK] * B[KxN], with B given K-major (Bt[N][K]).
// 128 threads = 4 waves; wave -> 32x64 (2x4 wmma accs); block -> 64x128.
// Tiles staged by async global->LDS b128, double buffered (ASYNCcnt).
// ---------------------------------------------------------------------------
enum { EP_BF16 = 0, EP_BF16T = 1, EP_F32SCALE = 2, EP_RESID = 3 };

template <int MODE>
__global__ void __launch_bounds__(128)
gemm_wmma(const __bf16* __restrict__ A, long sA,
          const __bf16* __restrict__ Bt, long sB,
          void* __restrict__ Dv, long sD,
          const float* __restrict__ bias,
          const float* __restrict__ X,
          float scale,
          int M, int N, int K, int lda, int ldb, int ldd) {
  __shared__ __bf16 As[2][64][40];   // [buf][row(M)][K], 80B rows (16B aligned)
  __shared__ __bf16 Bs[2][128][40];  // [buf][row(N)][K]

  const int b = blockIdx.z;
  const __bf16* Ab = A  + (size_t)b * sA;
  const __bf16* Bb = Bt + (size_t)b * sB;
  const int tile_m = blockIdx.y * 64;
  const int tile_n = blockIdx.x * 128;
  const int t = threadIdx.x;
  const int lane = t & 31;
  const int wave = t >> 5;
  const int wm = wave >> 1, wn = wave & 1;
  const int lm = lane & 15, hi = lane >> 4;

  // staging: A tile 64x32 -> 32B/thread ; B tile 128x32 -> 64B/thread
  const int arow = t >> 1;
  const int acol = (t & 1) * 16;

  v8f acc[2][4];
#pragma unroll
  for (int i = 0; i < 2; i++)
#pragma unroll
    for (int j = 0; j < 4; j++)
      acc[i][j] = (v8f){0.f, 0.f, 0.f, 0.f, 0.f, 0.f, 0.f, 0.f};

  const int T = K >> 5;

  auto stage = [&](int buf, int k0) {
    const __bf16* ga = Ab + (size_t)(tile_m + arow) * lda + k0 + acol;
    const unsigned la = lds_off32(&As[buf][arow][acol]);
    async_b128(la,      ga);
    async_b128(la + 16, ga + 8);
    const __bf16* gbp = Bb + (size_t)(tile_n + t) * ldb + k0;
    const unsigned lb = lds_off32(&Bs[buf][t][0]);
    async_b128(lb,      gbp);
    async_b128(lb + 16, gbp + 8);
    async_b128(lb + 32, gbp + 16);
    async_b128(lb + 48, gbp + 24);
  };

  // ---- stage tile 0 ----
  stage(0, 0);
  wait_async0();
  __syncthreads();

  for (int kt = 0; kt < T; ++kt) {
    const int cur = kt & 1;
    // prefetch next tile into the other buffer (overlaps with WMMA below)
    if (kt + 1 < T) stage(cur ^ 1, (kt + 1) << 5);

    // A fragments: lane = M row; elem chunks at byte {16*hi, 32+16*hi}
    v16bf af[2], bfv[4];
#pragma unroll
    for (int s = 0; s < 2; s++) {
      const int m = wm * 32 + s * 16 + lm;
      const u32x4* p = (const u32x4*)&As[cur][m][8 * hi];
      FragU u;
      u.q[0] = p[0];
      u.q[1] = p[2];
      af[s] = u.v;
    }
    // B fragments: lane = N col; elem chunks at byte {16*hi, 32+16*hi}
#pragma unroll
    for (int s = 0; s < 4; s++) {
      const int n = wn * 64 + s * 16 + lm;
      const u32x4* p = (const u32x4*)&Bs[cur][n][8 * hi];
      FragU u;
      u.q[0] = p[0];
      u.q[1] = p[2];
      bfv[s] = u.v;
    }

#pragma unroll
    for (int i = 0; i < 2; i++)
#pragma unroll
      for (int j = 0; j < 4; j++)
        acc[i][j] = __builtin_amdgcn_wmma_f32_16x16x32_bf16(
            false, af[i], false, bfv[j], (short)0, acc[i][j], false, false);

    if (kt + 1 < T) wait_async0();
    __syncthreads();
  }

  // Epilogue. C/D layout: VGPR r -> M = r + 8*hi ; N = lane&15.
#pragma unroll
  for (int i = 0; i < 2; i++) {
#pragma unroll
    for (int j = 0; j < 4; j++) {
      const int row0 = tile_m + wm * 32 + i * 16 + hi * 8;
      const int col  = tile_n + wn * 64 + j * 16 + lm;
#pragma unroll
      for (int r = 0; r < 8; r++) {
        const int row = row0 + r;
        const float v = acc[i][j][r];
        if (MODE == EP_BF16) {
          const float bv = bias ? bias[row] : 0.f;
          ((__bf16*)Dv)[(size_t)b * sD + (size_t)row * ldd + col] = (__bf16)(v + bv);
        } else if (MODE == EP_BF16T) {
          ((__bf16*)Dv)[(size_t)b * sD + (size_t)col * ldd + row] = (__bf16)(v + bias[row]);
        } else if (MODE == EP_F32SCALE) {
          ((float*)Dv)[(size_t)b * sD + (size_t)row * ldd + col] = v * scale;
        } else {  // EP_RESID: row = spatial n, col = out channel o
          const size_t idx = (size_t)b * sD + (size_t)col * ldd + row;
          ((float*)Dv)[idx] = X[idx] + v + bias[col];
        }
      }
    }
  }
}

// ---------------------------------------------------------------------------
extern "C" void kernel_launch(void* const* d_in, const int* in_sizes, int n_in,
                              void* d_out, int out_size, void* d_ws, size_t ws_size,
                              hipStream_t stream) {
  (void)in_sizes; (void)n_in; (void)out_size; (void)ws_size;
  const float* x   = (const float*)d_in[0];
  const float* gnw = (const float*)d_in[1];
  const float* gnb = (const float*)d_in[2];
  const float* wq  = (const float*)d_in[3];
  const float* bq  = (const float*)d_in[4];
  const float* wk  = (const float*)d_in[5];
  const float* bk  = (const float*)d_in[6];
  const float* wv  = (const float*)d_in[7];
  const float* bv  = (const float*)d_in[8];
  const float* wp  = (const float*)d_in[9];
  const float* bp  = (const float*)d_in[10];
  float* out = (float*)d_out;

  char* ws = (char*)d_ws;
  const size_t SZ_BCN = (size_t)NB * CCH * NSP * 2;  // 16 MB bf16
  const size_t SZ_W   = (size_t)CCH * CCH * 2;       // 512 KB
  size_t off = 0;
  __bf16* rT   = (__bf16*)(ws + off); off += SZ_BCN;  // [B][N][C]
  __bf16* qT   = (__bf16*)(ws + off); off += SZ_BCN;  // [B][N][C]
  __bf16* kT   = (__bf16*)(ws + off); off += SZ_BCN;  // [B][N][C]
  __bf16* v_b  = (__bf16*)(ws + off); off += SZ_BCN;  // [B][C][N]
  __bf16* wts  = (__bf16*)(ws + off); off += (size_t)NB * NSP * NSP * 2;  // 32 MB
  __bf16* wqb  = (__bf16*)(ws + off); off += SZ_W;
  __bf16* wkb  = (__bf16*)(ws + off); off += SZ_W;
  __bf16* wvb  = (__bf16*)(ws + off); off += SZ_W;
  __bf16* wpb  = (__bf16*)(ws + off); off += SZ_W;
  float*  sc   = (float*)(ws + off);                  // 64 MB f32 scores
  __bf16* outT = (__bf16*)(ws + off);                 // overlays scores (dead after softmax)

  const long sR = (long)CCH * NSP;

  gn_kernel<<<dim3(NB * 32), 256, 0, stream>>>(x, gnw, gnb, rT);
  wconv_kernel<<<dim3((CCH * CCH + 255) / 256), 256, 0, stream>>>(
      wq, wk, wv, wp, wqb, wkb, wvb, wpb);

  // Q = wq*r -> qT[n][o] (+bq)
  gemm_wmma<EP_BF16T><<<dim3(NSP / 128, CCH / 64, NB), 128, 0, stream>>>(
      wqb, 0, rT, sR, qT, sR, bq, nullptr, 1.f, CCH, NSP, CCH, CCH, CCH, CCH);
  // K = wk*r -> kT[n][o] (+bk)
  gemm_wmma<EP_BF16T><<<dim3(NSP / 128, CCH / 64, NB), 128, 0, stream>>>(
      wkb, 0, rT, sR, kT, sR, bk, nullptr, 1.f, CCH, NSP, CCH, CCH, CCH, CCH);
  // V = wv*r -> v_b[o][n] (+bv)   (plain layout == Bt for the AV GEMM)
  gemm_wmma<EP_BF16><<<dim3(NSP / 128, CCH / 64, NB), 128, 0, stream>>>(
      wvb, 0, rT, sR, v_b, sR, bv, nullptr, 1.f, CCH, NSP, CCH, CCH, CCH, NSP);
  // scores = qT * k / sqrt(C)  (A=qT[n][c], Bt=kT[j][c])
  gemm_wmma<EP_F32SCALE><<<dim3(NSP / 128, NSP / 64, NB), 128, 0, stream>>>(
      qT, sR, kT, sR, sc, (long)NSP * NSP, nullptr, nullptr,
      0.044194173824159216f, NSP, NSP, CCH, CCH, CCH, NSP);
  // softmax rows -> bf16 weights
  softmax_kernel<<<dim3(NB * NSP), 256, 0, stream>>>(sc, wts);
  // outT = wts * v   (A=wts[i][j], Bt=v_b[c][j]) -> outT[i][c]
  gemm_wmma<EP_BF16><<<dim3(CCH / 128, NSP / 64, NB), 128, 0, stream>>>(
      wts, (long)NSP * NSP, v_b, sR, outT, sR, nullptr, nullptr, 1.f,
      NSP, CCH, NSP, NSP, NSP, CCH);
  // y = x + outT * wp^T + bp   (A=outT[n][c], Bt=wpb[o][c]) -> out[b][o][n]
  gemm_wmma<EP_RESID><<<dim3(CCH / 128, NSP / 64, NB), 128, 0, stream>>>(
      outT, sR, wpb, 0, out, sR, bp, x, 1.f,
      NSP, CCH, CCH, CCH, CCH, NSP);
}
